// GCNBert_32160715112914
// MI455X (gfx1250) — compile-verified, matrix operand order checked by reading, not run on hardware
//
#include <hip/hip_runtime.h>
#include <math.h>

typedef __attribute__((ext_vector_type(16))) _Float16 v16h;
typedef __attribute__((ext_vector_type(8)))  float    v8f;
typedef __attribute__((ext_vector_type(4)))  int      v4i;
typedef __attribute__((address_space(1))) v4i as1_v4i;
typedef __attribute__((address_space(3))) v4i as3_v4i;
typedef __attribute__((address_space(3))) void as3v;

#define NNODES 5120
#define NB     128
#define NEDGE  20000
#define LPSEQ  512
#define MAXN   45
#define BSTRIDE 40   // LDS row stride in halves (80B: 16B-aligned, conflict-free)

// ---------------------------------------------------------------------------
// CDNA5 async b128 copy global->LDS (ASYNCcnt-tracked), with fallbacks.
// Builtin signature (from probe): (v4i AS1*, v4i AS3*, imm offset, imm cpol)
// ---------------------------------------------------------------------------
__device__ __forceinline__ void copy_b128_to_lds(const _Float16* g, _Float16* l) {
#if defined(__gfx1250__)
#if __has_builtin(__builtin_amdgcn_global_load_async_to_lds_b128)
    __builtin_amdgcn_global_load_async_to_lds_b128((as1_v4i*)g, (as3_v4i*)l, 0, 0);
#else
    unsigned loff = (unsigned)(uintptr_t)(as3v*)l;   // LDS byte offset
    asm volatile("global_load_async_to_lds_b128 %0, %1, off"
                 :: "v"(loff), "v"(g) : "memory");
#endif
#else
    *reinterpret_cast<uint4*>(l) = *reinterpret_cast<const uint4*>(g);
#endif
}
__device__ __forceinline__ void wait_async_copies() {
#if defined(__gfx1250__)
#if __has_builtin(__builtin_amdgcn_s_wait_asynccnt)
    __builtin_amdgcn_s_wait_asynccnt(0);
#else
    asm volatile("s_wait_asynccnt 0x0" ::: "memory");
#endif
#endif
}

// ---------------------------------------------------------------------------
// Repack f32 weight W[K][N] -> f16 Wp[(kt*N + n)*32 + k'] (WMMA B-fragment order)
// ---------------------------------------------------------------------------
__global__ void repack_w_kernel(const float* __restrict__ W, _Float16* __restrict__ Wp,
                                int K, int Nf) {
    int i = blockIdx.x * blockDim.x + threadIdx.x;
    if (i >= K * Nf) return;
    int kp   = i & 31;
    int rest = i >> 5;
    int n    = rest % Nf;
    int kt   = rest / Nf;
    Wp[i] = (_Float16)W[(size_t)(kt * 32 + kp) * Nf + n];
}

// ---------------------------------------------------------------------------
// WMMA f16->f32 GEMM:  C[M,N] = act(A[M,K] @ W[K,N] + bias)
// A row-major f32 (optional row gather: phys = gather[r/seg]*seg + r%seg).
// W pre-packed f16 (fragment order). 128 thr = 4 waves; wave = 16 rows x 64 cols.
// B tile (32x64, 4KB) staged via async-to-LDS, shared by all 4 waves.
// Requires M%64==0, N%64==0, K%32==0.
// ---------------------------------------------------------------------------
template<int ACT>
__global__ __launch_bounds__(128)
void wmma_gemm_kernel(const float* __restrict__ A, const _Float16* __restrict__ Wp,
                      const float* __restrict__ bias, float* __restrict__ C,
                      int M, int Nf, int K,
                      const int* __restrict__ gather, int gatherSeg)
{
    __shared__ _Float16 Bs[64 * BSTRIDE];
    const int tid  = threadIdx.x;
    const int lane = tid & 31;
    const int wv   = tid >> 5;
    const int h    = lane >> 4;        // lane half (0/1)
    const int r    = lane & 15;
    const int nbase = blockIdx.x * 64;
    const int mwave = blockIdx.y * 64 + wv * 16;

    const int mrow = mwave + r;        // A row this lane streams
    size_t arow;
    if (gather) arow = (size_t)gather[mrow / gatherSeg] * (size_t)gatherSeg
                       + (size_t)(mrow % gatherSeg);
    else        arow = (size_t)mrow;
    const float* __restrict__ Abase = A + arow * (size_t)K;

    // staging chunk assignment: 256 b128 chunks per tile, 2 per thread
    const int i0 = tid, i1 = 128 + tid;
    const int n0 = i0 >> 2, j0 = i0 & 3;
    const int n1 = i1 >> 2, j1 = i1 & 3;

    v8f acc[4] = {};

    for (int kk = 0; kk < K; kk += 32) {
        // ---- async-stage packed B tile (contiguous 4KB) into padded LDS rows ----
        const _Float16* __restrict__ Wt = Wp + ((size_t)(kk >> 5) * Nf + nbase) * 32;
        copy_b128_to_lds(Wt + i0 * 8, &Bs[n0 * BSTRIDE + j0 * 8]);
        copy_b128_to_lds(Wt + i1 * 8, &Bs[n1 * BSTRIDE + j1 * 8]);
        if (kk + 32 < K) __builtin_prefetch(Abase + kk + 32, 0, 3);
        wait_async_copies();
        __syncthreads();

        // ---- A fragment: 16-bit A 16x32 layout, K split by lane-half ----
        v16h afrag;
        const float* __restrict__ Ak = Abase + kk + h * 8;
        #pragma unroll
        for (int j = 0; j < 8; ++j) afrag[j]     = (_Float16)Ak[j];
        #pragma unroll
        for (int j = 0; j < 8; ++j) afrag[8 + j] = (_Float16)Ak[16 + j];

        // ---- 4 N-tiles per wave ----
        #pragma unroll
        for (int t = 0; t < 4; ++t) {
            v16h bfrag;
            const _Float16* __restrict__ Bp = &Bs[(t * 16 + r) * BSTRIDE + h * 8];
            #pragma unroll
            for (int j = 0; j < 8; ++j) bfrag[j]     = Bp[j];
            #pragma unroll
            for (int j = 0; j < 8; ++j) bfrag[8 + j] = Bp[16 + j];
            acc[t] = __builtin_amdgcn_wmma_f32_16x16x32_f16(
                         false, afrag, false, bfrag, (short)0, acc[t], false, false);
        }
        __syncthreads();
    }

    // ---- epilogue: C/D layout lane=n, VGPR q -> M = q + 8*half ----
    #pragma unroll
    for (int t = 0; t < 4; ++t) {
        int ncol = nbase + t * 16 + r;
        float bv = bias ? bias[ncol] : 0.0f;
        #pragma unroll
        for (int q = 0; q < 8; ++q) {
            int m = mwave + q + 8 * h;
            float v = acc[t][q] + bv;
            if (ACT) v = fmaxf(v, 0.0f);
            C[(size_t)m * Nf + ncol] = v;
        }
    }
}

// ---------------------------------------------------------------------------
// Small f32 tiled GEMM (for K=78 GCN layers etc.)
// ---------------------------------------------------------------------------
__global__ void gemm_f32_kernel(const float* __restrict__ A, const float* __restrict__ W,
                                float* __restrict__ C, int M, int Nf, int K)
{
    __shared__ float As[16][17], Wsm[16][17];
    int tx = threadIdx.x, ty = threadIdx.y;
    int row = blockIdx.y * 16 + ty;
    int col = blockIdx.x * 16 + tx;
    float acc = 0.0f;
    for (int k0 = 0; k0 < K; k0 += 16) {
        As[ty][tx]  = (row < M && (k0 + tx) < K) ? A[(size_t)row * K + k0 + tx] : 0.0f;
        Wsm[ty][tx] = ((k0 + ty) < K && col < Nf) ? W[(size_t)(k0 + ty) * Nf + col] : 0.0f;
        __syncthreads();
        #pragma unroll
        for (int kk = 0; kk < 16; ++kk) acc += As[ty][kk] * Wsm[kk][tx];
        __syncthreads();
    }
    if (row < M && col < Nf) C[(size_t)row * Nf + col] = acc;
}

// ---------------------------------------------------------------------------
// GCN helpers
// ---------------------------------------------------------------------------
__global__ void edge_count_kernel(const int* __restrict__ dst, float* deg, int ne) {
    int e = blockIdx.x * blockDim.x + threadIdx.x;
    if (e < ne) atomicAdd(&deg[dst[e]], 1.0f);
}
__global__ void dinv_kernel(const float* __restrict__ deg, float* dinv, int n) {
    int i = blockIdx.x * blockDim.x + threadIdx.x;
    if (i < n) dinv[i] = rsqrtf(deg[i] + 1.0f);
}
__global__ void gcn_self_kernel(const float* __restrict__ xw, const float* __restrict__ dinv,
                                const float* __restrict__ b, float* __restrict__ out,
                                int n, int F) {
    int i = blockIdx.x * blockDim.x + threadIdx.x;
    if (i >= n * F) return;
    int node = i / F, f = i % F;
    float dv = dinv[node];
    out[i] = xw[i] * dv * dv + b[f];
}
__global__ void gcn_edge_kernel(const int* __restrict__ src, const int* __restrict__ dst,
                                const float* __restrict__ xw, const float* __restrict__ dinv,
                                float* __restrict__ out, int ne, int F) {
    int i = blockIdx.x * blockDim.x + threadIdx.x;
    if (i >= ne * F) return;
    int e = i / F, f = i % F;
    int s = src[e], d = dst[e];
    atomicAdd(&out[(size_t)d * F + f], xw[(size_t)s * F + f] * dinv[s] * dinv[d]);
}
__global__ void relu_kernel(float* x, int n) {
    int i = blockIdx.x * blockDim.x + threadIdx.x;
    if (i < n) x[i] = fmaxf(x[i], 0.0f);
}

// ---------------------------------------------------------------------------
// to_dense_batch
// ---------------------------------------------------------------------------
__global__ void batch_count_kernel(const int* __restrict__ batch, int* counts, int n) {
    int i = blockIdx.x * blockDim.x + threadIdx.x;
    if (i < n) atomicAdd(&counts[batch[i]], 1);
}
__global__ void starts_kernel(const int* __restrict__ counts, int* starts, int nb) {
    if (blockIdx.x == 0 && threadIdx.x == 0) {
        int a = 0;
        for (int g = 0; g < nb; ++g) { starts[g] = a; a += counts[g]; }
    }
}
__global__ void scatter_dense_kernel(const float* __restrict__ hsrc, const int* __restrict__ batch,
                                     const int* __restrict__ starts, float* __restrict__ hd,
                                     int n, int F) {
    int i = blockIdx.x * blockDim.x + threadIdx.x;
    if (i >= n * F) return;
    int node = i / F, f = i % F;
    int g = batch[node];
    int pos = node - starts[g];
    if (pos >= 0 && pos < MAXN)
        hd[((size_t)g * MAXN + pos) * F + f] = hsrc[i];
}

// ---------------------------------------------------------------------------
// Co-attention
// ---------------------------------------------------------------------------
__global__ void coattn_C_kernel(const float* __restrict__ tb, const float* __restrict__ hd2,
                                float* __restrict__ Cm) {
    int b = blockIdx.x, tid = threadIdx.x;
    __shared__ float xcs[MAXN * 128];
    for (int i = tid; i < MAXN * 128; i += 256) xcs[i] = hd2[(size_t)b * MAXN * 128 + i];
    __syncthreads();
    int o = blockIdx.y * 256 + tid;
    if (o < LPSEQ * MAXN) {
        int l = o / MAXN, s = o % MAXN;
        const float* trow = tb + ((size_t)b * LPSEQ + l) * 128;
        const float* xrow = xcs + s * 128;
        float acc = 0.0f;
        #pragma unroll 4
        for (int m = 0; m < 128; ++m) acc += trow[m] * xrow[m];
        Cm[((size_t)b * LPSEQ + l) * MAXN + s] = tanhf(acc);
    }
}
__global__ void wcx_kernel(const float* __restrict__ W_c, const float* __restrict__ hd2,
                           float* __restrict__ Wcx) {
    int i = blockIdx.x * blockDim.x + threadIdx.x;
    if (i >= NB * 32 * MAXN) return;
    int s = i % MAXN, k = (i / MAXN) % 32, b = i / (MAXN * 32);
    const float* wr = W_c + k * 128;
    const float* xr = hd2 + ((size_t)b * MAXN + s) * 128;
    float acc = 0.0f;
    #pragma unroll 4
    for (int m = 0; m < 128; ++m) acc += wr[m] * xr[m];
    Wcx[i] = acc;
}
__global__ void wpt_kernel(const float* __restrict__ W_p, const float* __restrict__ t2,
                           float* __restrict__ Wpt) {
    int i = blockIdx.x * blockDim.x + threadIdx.x;
    if (i >= NB * 32 * LPSEQ) return;
    int l = i % LPSEQ, k = (i / LPSEQ) % 32, b = i / (LPSEQ * 32);
    const float* wr = W_p + k * 128;
    const float* tr = t2 + ((size_t)b * LPSEQ + l) * 128;
    float acc = 0.0f;
    #pragma unroll 4
    for (int m = 0; m < 128; ++m) acc += wr[m] * tr[m];
    Wpt[i] = acc;
}
__global__ void hc_kernel(const float* __restrict__ Wcx, const float* __restrict__ Wpt,
                          const float* __restrict__ Cm, const float* __restrict__ w_hc,
                          float* __restrict__ Hcw) {
    int i = blockIdx.x * blockDim.x + threadIdx.x;
    if (i >= NB * 32 * MAXN) return;
    int s = i % MAXN, k = (i / MAXN) % 32, b = i / (MAXN * 32);
    const float* wr = Wpt + ((size_t)b * 32 + k) * LPSEQ;
    const float* Cb = Cm + (size_t)b * LPSEQ * MAXN + s;
    float acc = Wcx[i];
    for (int l = 0; l < LPSEQ; ++l) acc += wr[l] * Cb[(size_t)l * MAXN];
    Hcw[i] = w_hc[k] * tanhf(acc);
}
__global__ void hp_kernel(const float* __restrict__ Wcx, const float* __restrict__ Wpt,
                          const float* __restrict__ Cm, const float* __restrict__ w_hp,
                          float* __restrict__ Hpw) {
    int i = blockIdx.x * blockDim.x + threadIdx.x;
    if (i >= NB * 32 * LPSEQ) return;
    int l = i % LPSEQ, k = (i / LPSEQ) % 32, b = i / (LPSEQ * 32);
    const float* xr = Wcx + ((size_t)b * 32 + k) * MAXN;
    const float* Cr = Cm + ((size_t)b * LPSEQ + l) * MAXN;
    float acc = Wpt[i];
    #pragma unroll
    for (int s = 0; s < MAXN; ++s) acc += xr[s] * Cr[s];
    Hpw[i] = w_hp[k] * tanhf(acc);
}
__global__ void reduce_k32_kernel(const float* __restrict__ src, float* __restrict__ dst, int L) {
    int i = blockIdx.x * blockDim.x + threadIdx.x;
    if (i >= NB * L) return;
    int b = i / L, j = i % L;
    float acc = 0.0f;
    #pragma unroll
    for (int k = 0; k < 32; ++k) acc += src[((size_t)b * 32 + k) * L + j];
    dst[i] = acc;
}
__global__ void softmax_kernel(const float* __restrict__ s, float* __restrict__ a, int L) {
    __shared__ float red[256];
    int b = blockIdx.x, tid = threadIdx.x;
    const float* row = s + (size_t)b * L;
    float* arow = a + (size_t)b * L;
    float mx = -1e30f;
    for (int i = tid; i < L; i += 256) mx = fmaxf(mx, row[i]);
    red[tid] = mx; __syncthreads();
    for (int st = 128; st > 0; st >>= 1) {
        if (tid < st) red[tid] = fmaxf(red[tid], red[tid + st]);
        __syncthreads();
    }
    float m = red[0]; __syncthreads();
    float sum = 0.0f;
    for (int i = tid; i < L; i += 256) { float e = expf(row[i] - m); arow[i] = e; sum += e; }
    red[tid] = sum; __syncthreads();
    for (int st = 128; st > 0; st >>= 1) {
        if (tid < st) red[tid] += red[tid + st];
        __syncthreads();
    }
    float inv = 1.0f / red[0];
    for (int i = tid; i < L; i += 256) arow[i] *= inv;
}
__global__ void pool_c_kernel(const float* __restrict__ ac, const float* __restrict__ hd2,
                              float* __restrict__ cp) {
    int i = blockIdx.x * blockDim.x + threadIdx.x;
    if (i >= NB * 128) return;
    int b = i / 128, m = i % 128;
    float acc = 0.0f;
    #pragma unroll
    for (int s = 0; s < MAXN; ++s) acc += ac[b * MAXN + s] * hd2[((size_t)b * MAXN + s) * 128 + m];
    cp[(size_t)b * 256 + m] = acc;
}
__global__ void pool_p_kernel(const float* __restrict__ ap, const float* __restrict__ t2,
                              float* __restrict__ cp) {
    int i = blockIdx.x * blockDim.x + threadIdx.x;
    if (i >= NB * 128) return;
    int b = i / 128, m = i % 128;
    float acc = 0.0f;
    for (int l = 0; l < LPSEQ; ++l) acc += ap[b * LPSEQ + l] * t2[((size_t)b * LPSEQ + l) * 128 + m];
    cp[(size_t)b * 256 + 128 + m] = acc;
}
__global__ void out_kernel(const float* __restrict__ m2, const float* __restrict__ oW,
                           const float* __restrict__ ob, float* __restrict__ y) {
    int b = blockIdx.x * blockDim.x + threadIdx.x;
    if (b >= NB) return;
    float acc = ob[0];
    for (int j = 0; j < 512; ++j) acc += m2[(size_t)b * 512 + j] * oW[j];
    y[b] = acc;
}

// ---------------------------------------------------------------------------
extern "C" void kernel_launch(void* const* d_in, const int* in_sizes, int n_in,
                              void* d_out, int out_size, void* d_ws, size_t ws_size,
                              hipStream_t stream)
{
    const float* x        = (const float*)d_in[0];
    const int*   edge     = (const int*)d_in[1];
    const int*   target   = (const int*)d_in[2];
    const int*   batch    = (const int*)d_in[3];
    const float* proteins = (const float*)d_in[4];
    const float* gW1 = (const float*)d_in[5];   const float* gb1 = (const float*)d_in[6];
    const float* gW2 = (const float*)d_in[7];   const float* gb2 = (const float*)d_in[8];
    const float* gW3 = (const float*)d_in[9];   const float* gb3 = (const float*)d_in[10];
    const float* fc1W = (const float*)d_in[11]; const float* fc1b = (const float*)d_in[12];
    const float* fc2W = (const float*)d_in[13]; const float* fc2b = (const float*)d_in[14];
    const float* b1W = (const float*)d_in[15];  const float* b1b = (const float*)d_in[16];
    const float* b2W = (const float*)d_in[17];  const float* b2b = (const float*)d_in[18];
    const float* W_b = (const float*)d_in[19];
    const float* W_c = (const float*)d_in[20];
    const float* W_p = (const float*)d_in[21];
    const float* w_hc = (const float*)d_in[22];
    const float* w_hp = (const float*)d_in[23];
    const float* c1W = (const float*)d_in[24];  const float* c1b = (const float*)d_in[25];
    const float* c2W = (const float*)d_in[26];  const float* c2b = (const float*)d_in[27];
    const float* oW = (const float*)d_in[28];   const float* ob = (const float*)d_in[29];
    float* out = (float*)d_out;

    const int* esrc = edge;
    const int* edst = edge + NEDGE;

    // ---- workspace layout (floats), ~130 MB ----
    float* ws = (float*)d_ws;
    size_t off = 0;
    auto alloc = [&](size_t n) -> float* {
        float* p = ws + off; off += (n + 63) & ~(size_t)63; return p;
    };
    float* deg    = alloc(NNODES);
    float* dinv   = alloc(NNODES);
    float* xw     = alloc((size_t)NNODES * 256);
    float* hA     = alloc((size_t)NNODES * 256);
    float* hB     = alloc((size_t)NNODES * 256);
    int*   counts = (int*)alloc(NB);
    int*   starts = (int*)alloc(NB);
    float* hd     = alloc((size_t)NB * MAXN * 256);      // dense batch, zeroed
    float* f1     = alloc((size_t)NB * MAXN * 1024);     // fc1 out; recycled below
    float* hd2    = alloc((size_t)NB * MAXN * 128);      // fc2 out (== xc transposed)
    float* t1     = alloc((size_t)NB * LPSEQ * 128);     // bert1 out; recycled as tb
    float* t2     = alloc((size_t)NB * LPSEQ * 128);     // bert2 out (== t)
    float* Hcw    = alloc((size_t)NB * 32 * MAXN);
    float* Hpw    = alloc((size_t)NB * 32 * LPSEQ);
    float* Wcx    = alloc((size_t)NB * 32 * MAXN);
    float* sc     = alloc((size_t)NB * MAXN);
    float* sp     = alloc((size_t)NB * LPSEQ);
    float* ac     = alloc((size_t)NB * MAXN);
    float* ap     = alloc((size_t)NB * LPSEQ);
    float* cp     = alloc((size_t)NB * 256);
    float* m1     = alloc((size_t)NB * 1024);
    float* m2     = alloc((size_t)NB * 512);
    // packed f16 weights (WMMA fragment order), ~2.75MB total
    _Float16* fc1Wp = (_Float16*)alloc(256 * 1024 / 2);
    _Float16* fc2Wp = (_Float16*)alloc(1024 * 128 / 2);
    _Float16* b1Wp  = (_Float16*)alloc(1280 * 128 / 2);
    _Float16* b2Wp  = (_Float16*)alloc(128 * 128 / 2);
    _Float16* WbWp  = (_Float16*)alloc(128 * 128 / 2);
    _Float16* c1Wp  = (_Float16*)alloc(256 * 1024 / 2);
    _Float16* c2Wp  = (_Float16*)alloc(1024 * 512 / 2);
    // overlays: f1 dead after fc2; t1 dead after bert2
    float* Cmat   = f1;                                  // B*512*45
    float* Wpt    = f1 + (size_t)NB * LPSEQ * MAXN;      // B*32*512 (fits in f1)
    float* tb     = t1;

    // ---- repack all WMMA weights to f16 fragment order ----
    auto repack = [&](const float* W, _Float16* Wp, int K, int Nf) {
        repack_w_kernel<<<(K * Nf + 255) / 256, 256, 0, stream>>>(W, Wp, K, Nf);
    };
    repack(fc1W, fc1Wp, 256, 1024);
    repack(fc2W, fc2Wp, 1024, 128);
    repack(b1W,  b1Wp,  1280, 128);
    repack(b2W,  b2Wp,  128, 128);
    repack(W_b,  WbWp,  128, 128);
    repack(c1W,  c1Wp,  256, 1024);
    repack(c2W,  c2Wp,  1024, 512);

    // ---- degrees & symmetric normalization ----
    (void)hipMemsetAsync(deg, 0, NNODES * sizeof(float), stream);
    edge_count_kernel<<<(NEDGE + 255) / 256, 256, 0, stream>>>(edst, deg, NEDGE);
    dinv_kernel<<<(NNODES + 255) / 256, 256, 0, stream>>>(deg, dinv, NNODES);

    // ---- GCN stack ----
    auto gcn_layer = [&](const float* in, const float* W, const float* b,
                         float* outp, int K, int F) {
        dim3 gg((F + 15) / 16, (NNODES + 15) / 16);
        gemm_f32_kernel<<<gg, dim3(16, 16), 0, stream>>>(in, W, xw, NNODES, F, K);
        gcn_self_kernel<<<(NNODES * F + 255) / 256, 256, 0, stream>>>(xw, dinv, b, outp, NNODES, F);
        gcn_edge_kernel<<<(NEDGE * F + 255) / 256, 256, 0, stream>>>(esrc, edst, xw, dinv, outp, NEDGE, F);
        relu_kernel<<<(NNODES * F + 255) / 256, 256, 0, stream>>>(outp, NNODES * F);
    };
    gcn_layer(x,  gW1, gb1, hA, 78,  128);
    gcn_layer(hA, gW2, gb2, hB, 128, 128);
    gcn_layer(hB, gW3, gb3, hA, 128, 256);   // final node feats [N,256] in hA

    // ---- to_dense_batch ----
    (void)hipMemsetAsync(counts, 0, NB * sizeof(int), stream);
    batch_count_kernel<<<(NNODES + 255) / 256, 256, 0, stream>>>(batch, counts, NNODES);
    starts_kernel<<<1, 1, 0, stream>>>(counts, starts, NB);
    (void)hipMemsetAsync(hd, 0, (size_t)NB * MAXN * 256 * sizeof(float), stream);
    scatter_dense_kernel<<<(NNODES * 256 + 255) / 256, 256, 0, stream>>>(hA, batch, starts, hd, NNODES, 256);

    // ---- dense MLP on graph branch (WMMA) ----
    wmma_gemm_kernel<1><<<dim3(1024 / 64, (NB * MAXN) / 64), 128, 0, stream>>>(
        hd, fc1Wp, fc1b, f1, NB * MAXN, 1024, 256, nullptr, 1);
    wmma_gemm_kernel<1><<<dim3(128 / 64, (NB * MAXN) / 64), 128, 0, stream>>>(
        f1, fc2Wp, fc2b, hd2, NB * MAXN, 128, 1024, nullptr, 1);

    // ---- protein branch (WMMA; bert1 has fused gather) ----
    const int MT = NB * LPSEQ;  // 65536
    wmma_gemm_kernel<1><<<dim3(128 / 64, MT / 64), 128, 0, stream>>>(
        proteins, b1Wp, b1b, t1, MT, 128, 1280, target, LPSEQ);
    wmma_gemm_kernel<1><<<dim3(128 / 64, MT / 64), 128, 0, stream>>>(
        t1, b2Wp, b2b, t2, MT, 128, 128, nullptr, 1);
    wmma_gemm_kernel<0><<<dim3(128 / 64, MT / 64), 128, 0, stream>>>(
        t2, WbWp, nullptr, tb, MT, 128, 128, nullptr, 1);   // tb = t @ W_b

    // ---- co-attention ----
    coattn_C_kernel<<<dim3(NB, (LPSEQ * MAXN + 255) / 256), 256, 0, stream>>>(tb, hd2, Cmat);
    wcx_kernel<<<(NB * 32 * MAXN + 255) / 256, 256, 0, stream>>>(W_c, hd2, Wcx);
    wpt_kernel<<<(NB * 32 * LPSEQ + 255) / 256, 256, 0, stream>>>(W_p, t2, Wpt);
    hc_kernel<<<(NB * 32 * MAXN + 255) / 256, 256, 0, stream>>>(Wcx, Wpt, Cmat, w_hc, Hcw);
    hp_kernel<<<(NB * 32 * LPSEQ + 255) / 256, 256, 0, stream>>>(Wcx, Wpt, Cmat, w_hp, Hpw);
    reduce_k32_kernel<<<(NB * MAXN + 255) / 256, 256, 0, stream>>>(Hcw, sc, MAXN);
    reduce_k32_kernel<<<(NB * LPSEQ + 255) / 256, 256, 0, stream>>>(Hpw, sp, LPSEQ);
    softmax_kernel<<<NB, 256, 0, stream>>>(sc, ac, MAXN);
    softmax_kernel<<<NB, 256, 0, stream>>>(sp, ap, LPSEQ);
    pool_c_kernel<<<(NB * 128 + 255) / 256, 256, 0, stream>>>(ac, hd2, cp);
    pool_p_kernel<<<(NB * 128 + 255) / 256, 256, 0, stream>>>(ap, t2, cp);

    // ---- head MLP (WMMA) + output ----
    wmma_gemm_kernel<1><<<dim3(1024 / 64, NB / 64), 128, 0, stream>>>(
        cp, c1Wp, c1b, m1, NB, 1024, 256, nullptr, 1);
    wmma_gemm_kernel<1><<<dim3(512 / 64, NB / 64), 128, 0, stream>>>(
        m1, c2Wp, c2b, m2, NB, 512, 1024, nullptr, 1);
    out_kernel<<<1, 128, 0, stream>>>(m2, oW, ob, out);
}
